// LSTM_30416958390527
// MI455X (gfx1250) — matrix operand order repaired and checked by reading, not compiled, not run on hardware
//
#include <hip/hip_runtime.h>

// ---------------------------------------------------------------------------
// Fused 2-layer LSTM (IN=13, H=32, B=4096, T=512) + FC head for gfx1250.
// One wave32 owns a 16-row batch tile and runs the full recurrence of both
// layers with v_wmma_f32_16x16x32_f16. Weight B-fragments are staged through
// LDS once and hoisted into the 1024-VGPR file by the compiler; h state
// round-trips through a per-wave LDS buffer (D-layout -> A-layout).
// Nonlinearities use native v_tanh_f32 when available.
// ---------------------------------------------------------------------------

typedef _Float16 f16_t;
typedef __attribute__((ext_vector_type(16))) _Float16 v16h;
typedef __attribute__((ext_vector_type(8)))  float    v8f;
typedef __attribute__((ext_vector_type(4)))  unsigned u32x4;

#define IN_DIM 13
#define HID    32
#define BATCH  4096
#define SEQ    512
#define WAVES_PER_BLOCK 4
#define TILE_M 16

union FragU { v16h v; u32x4 q[2]; };

#if __has_builtin(__builtin_amdgcn_tanhf)
__device__ __forceinline__ float tanhx(float x) {
    return __builtin_amdgcn_tanhf(x);           // native v_tanh_f32 (1 TRANS op)
}
__device__ __forceinline__ float sigf(float x) {
    return __builtin_fmaf(0.5f, __builtin_amdgcn_tanhf(0.5f * x), 0.5f);
}
#else
__device__ __forceinline__ float sigf(float x) {
    return 1.0f / (1.0f + __expf(-x));
}
__device__ __forceinline__ float tanhx(float x) {
    return 2.0f * sigf(2.0f * x) - 1.0f;
}
#endif

__global__ __launch_bounds__(128) void lstm_wmma_kernel(
    const float* __restrict__ x,
    const float* __restrict__ W_ih0, const float* __restrict__ W_hh0,
    const float* __restrict__ b_ih0, const float* __restrict__ b_hh0,
    const float* __restrict__ W_ih1, const float* __restrict__ W_hh1,
    const float* __restrict__ b_ih1, const float* __restrict__ b_hh1,
    const float* __restrict__ fc_W,  const float* __restrict__ fc_b,
    float* __restrict__ out)
{
    // Weights in WMMA B-operand fragment layout, f16:
    //   wlds[mat][ntile][lane][h]  ->  element (K = (h&7) + 8*(lane>>4) + 16*(h>>3),
    //                                           N = ntile*16 + (lane&15))
    // mat: 0=W_ih0 (K padded 13->32 with zeros), 1=W_hh0, 2=W_ih1, 3=W_hh1
    __shared__ __attribute__((aligned(32))) f16_t wlds[4][8][32][16];   // 32 KB
    // Per-wave h staging buffers, row-major [m][j], f16 (D->A relayout).
    __shared__ __attribute__((aligned(16))) f16_t hbuf[WAVES_PER_BLOCK][2][16][HID]; // 8 KB

    const int tid    = threadIdx.x;
    const int wave   = tid >> 5;
    const int lane   = tid & 31;
    const int laneM  = lane & 15;   // col within C/D tile, row within A
    const int laneHi = lane >> 4;   // lane-group select

    // ---- one-time: stage weights into LDS fragment layout ----
    for (int idx = tid; idx < 4 * 8 * 32 * 16; idx += blockDim.x) {
        int mat = idx >> 12;
        int rem = idx & 4095;
        int nt  = rem >> 9;
        int r2  = rem & 511;
        int L   = r2 >> 4;
        int h   = r2 & 15;
        int N = nt * 16 + (L & 15);
        int K = (h & 7) + 8 * (L >> 4) + 16 * (h >> 3);
        float v;
        if (mat == 0)      v = (K < IN_DIM) ? W_ih0[N * IN_DIM + K] : 0.0f;
        else if (mat == 1) v = W_hh0[N * HID + K];
        else if (mat == 2) v = W_ih1[N * HID + K];
        else               v = W_hh1[N * HID + K];
        wlds[mat][nt][L][h] = (f16_t)v;
    }
    // zero initial h state
    for (int idx = tid; idx < WAVES_PER_BLOCK * 2 * 16 * HID; idx += blockDim.x) {
        ((f16_t*)hbuf)[idx] = (f16_t)0.0f;
    }
    __syncthreads();

    // bias (D-layout value depends only on N = nt*16 + laneM)
    float bias0[8], bias1[8];
    #pragma unroll
    for (int nt = 0; nt < 8; ++nt) {
        int n = nt * 16 + laneM;
        bias0[nt] = b_ih0[n] + b_hh0[n];
        bias1[nt] = b_ih1[n] + b_hh1[n];
    }

    const int b0 = (blockIdx.x * WAVES_PER_BLOCK + wave) * TILE_M;
    const float* xrow = x + (long)(b0 + laneM) * SEQ * IN_DIM;

    // cell state in D-fragment register layout: c[tt*8+r] is element
    // (m = r + 8*laneHi, j = tt*16 + laneM)
    float c0[16], c1[16];
    #pragma unroll
    for (int i = 0; i < 16; ++i) { c0[i] = 0.0f; c1[i] = 0.0f; }

    f16_t (*h0buf)[HID] = hbuf[wave][0];
    f16_t (*h1buf)[HID] = hbuf[wave][1];

    #pragma unroll 1
    for (int t = 0; t < SEQ; ++t) {
        // speculative prefetch of the x stream one cacheline ahead
        if (t + 8 < SEQ) __builtin_prefetch(&xrow[(t + 8) * IN_DIM], 0, 0);

        // ================= layer 0 =================
        // A operand from x (K padded 13 -> 32 with zeros). A layout:
        // lane L holds row M=L%16, half h -> K = (h&7) + 8*(L/16) + 16*(h>>3)
        FragU ax;
        #pragma unroll
        for (int h = 0; h < 16; ++h) {
            int K = (h & 7) + 8 * laneHi + 16 * (h >> 3);
            float v = 0.0f;
            if (K < IN_DIM) v = xrow[t * IN_DIM + K];
            ax.v[h] = (f16_t)v;
        }
        // A operand from h0: two contiguous 16B runs per lane
        FragU ah0;
        ah0.q[0] = *reinterpret_cast<const u32x4*>(&h0buf[laneM][laneHi * 8]);
        ah0.q[1] = *reinterpret_cast<const u32x4*>(&h0buf[laneM][16 + laneHi * 8]);

        v8f g0[8];
        #pragma unroll
        for (int nt = 0; nt < 8; ++nt) {
            float b = bias0[nt];
            v8f acc = {b, b, b, b, b, b, b, b};
            v16h bih = *reinterpret_cast<const v16h*>(&wlds[0][nt][lane][0]);
            acc = __builtin_amdgcn_wmma_f32_16x16x32_f16(false, ax.v,  false, bih,
                                                         (short)0, acc, false, false);
            v16h bhh = *reinterpret_cast<const v16h*>(&wlds[1][nt][lane][0]);
            acc = __builtin_amdgcn_wmma_f32_16x16x32_f16(false, ah0.v, false, bhh,
                                                         (short)0, acc, false, false);
            g0[nt] = acc;
        }
        // cell update; gate tiles: i=(0,1) f=(2,3) g=(4,5) o=(6,7)
        #pragma unroll
        for (int tt = 0; tt < 2; ++tt) {
            #pragma unroll
            for (int r = 0; r < 8; ++r) {
                float ig = sigf(g0[0 + tt][r]);
                float fg = sigf(g0[2 + tt][r]);
                float gg = tanhx(g0[4 + tt][r]);
                float og = sigf(g0[6 + tt][r]);
                float c  = fg * c0[tt * 8 + r] + ig * gg;
                c0[tt * 8 + r] = c;
                float hv = og * tanhx(c);
                h0buf[r + 8 * laneHi][tt * 16 + laneM] = (f16_t)hv;
            }
        }

        // ================= layer 1 =================
        FragU axh;  // freshly written h0 is layer-1 input
        axh.q[0] = *reinterpret_cast<const u32x4*>(&h0buf[laneM][laneHi * 8]);
        axh.q[1] = *reinterpret_cast<const u32x4*>(&h0buf[laneM][16 + laneHi * 8]);
        FragU ah1;
        ah1.q[0] = *reinterpret_cast<const u32x4*>(&h1buf[laneM][laneHi * 8]);
        ah1.q[1] = *reinterpret_cast<const u32x4*>(&h1buf[laneM][16 + laneHi * 8]);

        v8f g1[8];
        #pragma unroll
        for (int nt = 0; nt < 8; ++nt) {
            float b = bias1[nt];
            v8f acc = {b, b, b, b, b, b, b, b};
            v16h bih = *reinterpret_cast<const v16h*>(&wlds[2][nt][lane][0]);
            acc = __builtin_amdgcn_wmma_f32_16x16x32_f16(false, axh.v, false, bih,
                                                         (short)0, acc, false, false);
            v16h bhh = *reinterpret_cast<const v16h*>(&wlds[3][nt][lane][0]);
            acc = __builtin_amdgcn_wmma_f32_16x16x32_f16(false, ah1.v, false, bhh,
                                                         (short)0, acc, false, false);
            g1[nt] = acc;
        }
        #pragma unroll
        for (int tt = 0; tt < 2; ++tt) {
            #pragma unroll
            for (int r = 0; r < 8; ++r) {
                float ig = sigf(g1[0 + tt][r]);
                float fg = sigf(g1[2 + tt][r]);
                float gg = tanhx(g1[4 + tt][r]);
                float og = sigf(g1[6 + tt][r]);
                float c  = fg * c1[tt * 8 + r] + ig * gg;
                c1[tt * 8 + r] = c;
                float hv = og * tanhx(c);
                h1buf[r + 8 * laneHi][tt * 16 + laneM] = (f16_t)hv;
            }
        }
    }

    // ---- FC head: out[b0+m] = fc_b + sum_j h1_T[m][j] * fc_W[j] ----
    if (lane < 16) {
        float s = fc_b[0];
        #pragma unroll
        for (int j = 0; j < HID; ++j) {
            s += (float)h1buf[lane][j] * fc_W[j];
        }
        out[b0 + lane] = s;
    }
}

extern "C" void kernel_launch(void* const* d_in, const int* in_sizes, int n_in,
                              void* d_out, int out_size, void* d_ws, size_t ws_size,
                              hipStream_t stream) {
    const float* x     = (const float*)d_in[0];
    const float* W_ih0 = (const float*)d_in[1];
    const float* W_hh0 = (const float*)d_in[2];
    const float* b_ih0 = (const float*)d_in[3];
    const float* b_hh0 = (const float*)d_in[4];
    const float* W_ih1 = (const float*)d_in[5];
    const float* W_hh1 = (const float*)d_in[6];
    const float* b_ih1 = (const float*)d_in[7];
    const float* b_hh1 = (const float*)d_in[8];
    const float* fc_W  = (const float*)d_in[9];
    const float* fc_b  = (const float*)d_in[10];
    float* out = (float*)d_out;

    dim3 grid(BATCH / (TILE_M * WAVES_PER_BLOCK));  // 64 workgroups
    dim3 block(WAVES_PER_BLOCK * 32);               // 4 wave32s
    lstm_wmma_kernel<<<grid, block, 0, stream>>>(
        x, W_ih0, W_hh0, b_ih0, b_hh0,
        W_ih1, W_hh1, b_ih1, b_hh1, fc_W, fc_b, out);
}
